// VQ_SEEDLM_v2_31533649887670
// MI455X (gfx1250) — matrix operand order repaired and checked by reading, not compiled, not run on hardware
//
#include <hip/hip_runtime.h>
#include <hip/hip_bf16.h>
#include <math.h>

// ---------------------------------------------------------------------------
// Problem constants (match reference)
// ---------------------------------------------------------------------------
#define BB   16384            // batch of weight blocks
#define INN  64               // input dim
#define DENC 512              // encoder dim
#define NE   2048             // codebook entries
#define PP   12               // basis vectors per block
#define ZD   (INN*PP)         // 768
#define BP   (BB*PP)          // 196608

typedef __attribute__((ext_vector_type(16))) _Float16 v16h;
typedef __attribute__((ext_vector_type(8)))  _Float16 v8h;
typedef __attribute__((ext_vector_type(8)))  float    v8f;

// ---------------------------------------------------------------------------
// WMMA helper (CDNA5 wave32, 16x16x32 f16 -> f32 accumulate)
// ---------------------------------------------------------------------------
static __device__ __forceinline__ v8f wmma_f16(v16h a, v16h b, v8f c) {
  return __builtin_amdgcn_wmma_f32_16x16x32_f16(
      /*neg_a=*/false, a, /*neg_b=*/false, b,
      /*c_mod=*/(short)0, c, /*reuse_a=*/false, /*reuse_b=*/false);
}

// ---------------------------------------------------------------------------
// f16 fragment loaders: each fragment is two contiguous 16-byte chunks/lane
// (emitted as 2x global_load_b128 / ds_load_b128).
//
// A (16x32, ISA 7.12.2): lane L -> M=L%16, h=L/16;
//   elems 0..7  = K k0+8h   .. k0+8h+7
//   elems 8..15 = K k0+16+8h.. k0+16+8h+7
// B (32x16) from N-major (transposed) storage Bt[n][k]: lane L -> N=L%16;
//   elems 0..7  = K k0+8h   .. k0+8h+7
//   elems 8..15 = K k0+16+8h.. k0+16+8h+7
// ---------------------------------------------------------------------------
static __device__ __forceinline__ v16h cat8(v8h lo, v8h hi) {
  v16h r;
#pragma unroll
  for (int i = 0; i < 8; ++i) { r[i] = lo[i]; r[i + 8] = hi[i]; }
  return r;
}

static __device__ __forceinline__ v16h load_afrag_h(const _Float16* A, int ld,
                                                    int m0, int k0) {
  int lane = threadIdx.x & 31;
  const _Float16* row =
      A + (size_t)(m0 + (lane & 15)) * ld + k0 + 8 * (lane >> 4);
  return cat8(*(const v8h*)row, *(const v8h*)(row + 16));
}

static __device__ __forceinline__ v16h load_bfrag_h(const _Float16* Bt, int ld,
                                                    int n0, int k0) {
  int lane = threadIdx.x & 31;
  const _Float16* row =
      Bt + (size_t)(n0 + (lane & 15)) * ld + k0 + 8 * (lane >> 4);
  return cat8(*(const v8h*)row, *(const v8h*)(row + 16));
}

// ---------------------------------------------------------------------------
// Async global -> LDS staging (CDNA5 GLOBAL_LOAD_ASYNC_TO_LDS, ASYNCcnt)
// ---------------------------------------------------------------------------
static __device__ __forceinline__ void async_b128(unsigned lds_addr,
                                                  const void* gptr) {
  asm volatile("global_load_async_to_lds_b128 %0, %1, off"
               :: "v"(lds_addr), "v"((unsigned long long)(uintptr_t)gptr)
               : "memory");
}
static __device__ __forceinline__ void wait_async_le4() {
  asm volatile("s_wait_asynccnt 0x4" ::: "memory");
}
static __device__ __forceinline__ void wait_async_0() {
  asm volatile("s_wait_asynccnt 0x0" ::: "memory");
}

// ---------------------------------------------------------------------------
// Kernel 0: f16 conversions (+ weight transposes), 0.5*||e||^2, zero accums.
// Runs every launch (graph-replay safe).
// ---------------------------------------------------------------------------
__global__ void convert_kernel(const float* x, const float* w1, const float* wr,
                               const float* w2, const float* emb, _Float16* xh,
                               _Float16* w1t, _Float16* wrt, _Float16* w2t,
                               _Float16* embh, float* en2h, float* counts,
                               float* lossacc) {
  int t = blockIdx.x * blockDim.x + threadIdx.x;
  int stride = gridDim.x * blockDim.x;
  for (int i = t; i < BB * INN; i += stride) xh[i] = (_Float16)x[i];
  for (int i = t; i < DENC * INN; i += stride) {   // w1 (64x512) -> w1t (512x64)
    int n = i / INN, k = i % INN;
    w1t[i] = (_Float16)w1[k * DENC + n];
  }
  for (int i = t; i < DENC * DENC; i += stride) {  // wr (512x512) -> wrt^T
    int n = i / DENC, k = i % DENC;
    wrt[i] = (_Float16)wr[k * DENC + n];
  }
  for (int i = t; i < ZD * DENC; i += stride) {    // w2 (512x768) -> w2t (768x512)
    int n = i / DENC, k = i % DENC;
    w2t[i] = (_Float16)w2[k * ZD + n];
  }
  for (int i = t; i < NE * INN; i += stride) embh[i] = (_Float16)emb[i];
  for (int n = t; n < NE; n += stride) {
    const float* e = emb + (size_t)n * INN;
    float s = 0.f;
    for (int i = 0; i < INN; ++i) s += e[i] * e[i];
    en2h[n] = 0.5f * s;
    counts[n] = 0.f;
  }
  if (t == 0) *lossacc = 0.f;
}

// ---------------------------------------------------------------------------
// GEMM + bias, dual f32/f16 output: C = A[MxK] @ Bt^T + bias
// grid = (M/16, N/128), block = 256 (8 waves, one 16x16 tile per wave)
// ---------------------------------------------------------------------------
__global__ void gemm_bias_kernel(const _Float16* A, const _Float16* Bt,
                                 const float* bias, float* Cf, _Float16* Ch,
                                 int N, int K) {
  int w  = threadIdx.x >> 5;
  int m0 = blockIdx.x * 16;
  int n0 = blockIdx.y * 128 + w * 16;
  v8f acc = {};
  for (int k0 = 0; k0 < K; k0 += 32) {
    v16h a = load_afrag_h(A, K, m0, k0);
    v16h b = load_bfrag_h(Bt, K, n0, k0);
    acc = wmma_f16(a, b, acc);
  }
  int lane = threadIdx.x & 31;
  int n = n0 + (lane & 15), hl = lane >> 4;
  float bv = bias[n];
#pragma unroll
  for (int r = 0; r < 8; ++r) {
    int m = m0 + r + 8 * hl;
    float v = acc[r] + bv;
    Cf[(size_t)m * N + n] = v;
    Ch[(size_t)m * N + n] = (_Float16)v;
  }
}

// ---------------------------------------------------------------------------
// Fused resblock: h += relu(layernorm(h @ W + b) * g + beta)
// One block owns 16 complete rows; 8 waves x 4 n-chunks; 64 WMMA / wave.
// ---------------------------------------------------------------------------
__global__ void resblock_kernel(float* hbuf, _Float16* hh, const _Float16* Wt,
                                const float* bias, const float* g,
                                const float* beta) {
  __shared__ float rt[16 * (DENC + 8)];
  int tid = threadIdx.x;
  int w = tid >> 5, lane = tid & 31;
  int m0 = blockIdx.x * 16;

  v8f acc[4] = {};
  for (int k0 = 0; k0 < DENC; k0 += 32) {
    v16h a = load_afrag_h(hh, DENC, m0, k0);
#pragma unroll
    for (int s = 0; s < 4; ++s) {
      int n0 = (s * 8 + w) * 16;
      v16h b = load_bfrag_h(Wt, DENC, n0, k0);
      acc[s] = wmma_f16(a, b, acc[s]);
    }
  }
  int hl = lane >> 4, nn = lane & 15;
#pragma unroll
  for (int s = 0; s < 4; ++s) {
    int n = (s * 8 + w) * 16 + nn;
    float bv = bias[n];
#pragma unroll
    for (int r = 0; r < 8; ++r)
      rt[(r + 8 * hl) * (DENC + 8) + n] = acc[s][r] + bv;
  }
  __syncthreads();

  // LayerNorm + ReLU + residual: 16 threads/row, shfl width-16 reductions
  int row = tid >> 4;
  int cl  = tid & 15;
  const float* rrow = &rt[row * (DENC + 8)];
  float s1 = 0.f, s2 = 0.f;
  for (int j = 0; j < DENC / 16; ++j) {
    float v = rrow[cl + j * 16];
    s1 += v; s2 += v * v;
  }
#pragma unroll
  for (int off = 1; off < 16; off <<= 1) {
    s1 += __shfl_xor(s1, off, 16);
    s2 += __shfl_xor(s2, off, 16);
  }
  float mu  = s1 * (1.f / DENC);
  float var = s2 * (1.f / DENC) - mu * mu;
  float inv = rsqrtf(var + 1e-5f);
  for (int j = 0; j < DENC / 16; ++j) {
    int c = cl + j * 16;
    float v = (rrow[c] - mu) * inv * g[c] + beta[c];
    v = fmaxf(v, 0.f);
    size_t gi = (size_t)(m0 + row) * DENC + c;
    float hv = hbuf[gi] + v;
    hbuf[gi] = hv;
    hh[gi]   = (_Float16)hv;
  }
}

// ---------------------------------------------------------------------------
// Codebook argmin over 0.5||e||^2 - z.e, 16 zf rows / block, 2048 codes.
// Codebook tiles (128 codes x 64 f16 = 16KB) are double-buffered in LDS via
// async global->LDS copies; B fragments then come from LDS (ds_load_b128).
// ---------------------------------------------------------------------------
__global__ void argmin_kernel(const _Float16* zh, const _Float16* embh,
                              const float* en2h, int* idx_out,
                              float* idx_outf) {
  __shared__ _Float16 tile[2][128 * INN];     // 2 x 16 KB
  __shared__ float sbest[8][16];
  __shared__ int   sidx[8][16];
  int tid = threadIdx.x, w = tid >> 5, lane = tid & 31;
  int m0 = blockIdx.x * 16;
  int nn = lane & 15, hl = lane >> 4;

  v16h a0 = load_afrag_h(zh, INN, m0, 0);
  v16h a1 = load_afrag_h(zh, INN, m0, 32);

  float best[8];
  int   bi[8];
#pragma unroll
  for (int r = 0; r < 8; ++r) { best[r] = 3.4e38f; bi[r] = 0; }

  // issue tile 0 (4 async b128 per thread = 16KB per block)
  {
    unsigned lb = (unsigned)(uintptr_t)(&tile[0][0]);
    const char* gp = (const char*)embh;
#pragma unroll
    for (int j = 0; j < 4; ++j) {
      unsigned o = (unsigned)(tid + j * 256) * 16u;
      async_b128(lb + o, gp + o);
    }
  }

  for (int s = 0; s < NE / 128; ++s) {
    int cur = s & 1;
    if (s + 1 < NE / 128) {
      unsigned lb = (unsigned)(uintptr_t)(&tile[1 - cur][0]);
      const char* gp = (const char*)(embh + (size_t)(s + 1) * 128 * INN);
#pragma unroll
      for (int j = 0; j < 4; ++j) {
        unsigned o = (unsigned)(tid + j * 256) * 16u;
        async_b128(lb + o, gp + o);
      }
      wait_async_le4();          // ASYNCcnt in-order: tile s resident
    } else {
      wait_async_0();
    }
    __syncthreads();             // tile visible to all waves

    v16h b0 = load_bfrag_h(&tile[cur][0], INN, w * 16, 0);
    v16h b1 = load_bfrag_h(&tile[cur][0], INN, w * 16, 32);
    v8f acc = {};
    acc = wmma_f16(a0, b0, acc);
    acc = wmma_f16(a1, b1, acc);

    int n = s * 128 + w * 16 + nn;
    float e2 = en2h[n];
#pragma unroll
    for (int r = 0; r < 8; ++r) {
      float cost = e2 - acc[r];
      if (cost < best[r]) { best[r] = cost; bi[r] = n; }
    }
    __syncthreads();             // done reading tile[cur] before re-fill
  }

  // argmin reduce across the 16 lanes covering the same rows
#pragma unroll
  for (int off = 1; off < 16; off <<= 1) {
#pragma unroll
    for (int r = 0; r < 8; ++r) {
      float ob = __shfl_xor(best[r], off, 16);
      int   oi = __shfl_xor(bi[r], off, 16);
      if (ob < best[r] || (ob == best[r] && oi < bi[r])) {
        best[r] = ob; bi[r] = oi;
      }
    }
  }
  if (nn == 0) {
#pragma unroll
    for (int r = 0; r < 8; ++r) {
      sbest[w][8 * hl + r] = best[r];
      sidx[w][8 * hl + r]  = bi[r];
    }
  }
  __syncthreads();
  if (tid < 16) {
    float bb = sbest[0][tid];
    int   ii = sidx[0][tid];
    for (int ww = 1; ww < 8; ++ww) {
      float ob = sbest[ww][tid];
      int   oi = sidx[ww][tid];
      if (ob < bb || (ob == bb && oi < ii)) { bb = ob; ii = oi; }
    }
    idx_out[m0 + tid]  = ii;
    idx_outf[m0 + tid] = (float)ii;
  }
}

// ---------------------------------------------------------------------------
// Loss sum + histogram counts (f32 path for accuracy)
// ---------------------------------------------------------------------------
__global__ void loss_counts_kernel(const float* zf, const float* emb,
                                   const int* idx, float* lossacc,
                                   float* counts) {
  __shared__ float red[256];
  int t = blockIdx.x * blockDim.x + threadIdx.x;
  float s = 0.f;
  for (int row = t; row < BP; row += gridDim.x * blockDim.x) {
    int id = idx[row];
    const float* e  = emb + (size_t)id * INN;
    const float* zr = zf + (size_t)row * INN;
    float rs = 0.f;
    for (int i = 0; i < INN; ++i) { float d = e[i] - zr[i]; rs += d * d; }
    s += rs;
    atomicAdd(&counts[id], 1.0f);
  }
  red[threadIdx.x] = s;
  __syncthreads();
  for (int off = 128; off > 0; off >>= 1) {
    if (threadIdx.x < off) red[threadIdx.x] += red[threadIdx.x + off];
    __syncthreads();
  }
  if (threadIdx.x == 0) atomicAdd(lossacc, red[0]);
}

// ---------------------------------------------------------------------------
// Finalize scalars: loss (1.25 * mse) and perplexity
// ---------------------------------------------------------------------------
__global__ void finalize_kernel(const float* lossacc, const float* counts,
                                float* out_loss, float* out_perp) {
  __shared__ float red[256];
  float s = 0.f;
  for (int n = threadIdx.x; n < NE; n += 256) {
    float c = counts[n] * (1.0f / (float)BP);
    s += c * logf(c + 1e-10f);
  }
  red[threadIdx.x] = s;
  __syncthreads();
  for (int off = 128; off > 0; off >>= 1) {
    if (threadIdx.x < off) red[threadIdx.x] += red[threadIdx.x + off];
    __syncthreads();
  }
  if (threadIdx.x == 0) {
    *out_perp = expf(-red[0]);
    *out_loss = 1.25f * lossacc[0] / ((float)BP * (float)INN);
  }
}

// ---------------------------------------------------------------------------
// Per-block pinv via normal equations: G = A^T A, b = A^T x, solve, x_hat
// ---------------------------------------------------------------------------
__global__ void pinv_kernel(const float* x, const float* emb, const int* idx,
                            float* out_xhat, float* out_coeff) {
  __shared__ float As[INN][PP + 1];
  __shared__ float xs[INN];
  __shared__ float G[PP][PP + 1];
  __shared__ float cf[PP];
  __shared__ int   sid[PP];
  int b = blockIdx.x, t = threadIdx.x;

  if (t < PP) sid[t] = idx[b * PP + t];
  __syncthreads();
  if (t < INN) {
    xs[t] = x[(size_t)b * INN + t];
    for (int p = 0; p < PP; ++p)
      As[t][p] = emb[(size_t)sid[p] * INN + t];
  }
  __syncthreads();

  if (t < 78) {                       // upper triangle of G (78 pairs)
    int p = 0, tt = t;
    while (tt >= PP - p) { tt -= PP - p; ++p; }
    int q = p + tt;
    float s = 0.f;
    for (int i = 0; i < INN; ++i) s += As[i][p] * As[i][q];
    G[p][q] = s; G[q][p] = s;
  } else if (t < 78 + PP) {           // rhs b = A^T x
    int p = t - 78;
    float s = 0.f;
    for (int i = 0; i < INN; ++i) s += As[i][p] * xs[i];
    G[p][PP] = s;
  }
  __syncthreads();

  if (t == 0) {                       // small serial Gaussian elimination
    for (int k = 0; k < PP; ++k) {
      float piv = G[k][k];
      if (fabsf(piv) < 1e-20f) piv = (piv >= 0.f) ? 1e-20f : -1e-20f;
      float inv = 1.f / piv;
      for (int j = k; j <= PP; ++j) G[k][j] *= inv;
      for (int r = k + 1; r < PP; ++r) {
        float f = G[r][k];
        for (int j = k; j <= PP; ++j) G[r][j] -= f * G[k][j];
      }
    }
    for (int k = PP - 1; k >= 0; --k) {
      float c = G[k][PP];
      for (int j = k + 1; j < PP; ++j) c -= G[k][j] * cf[j];
      cf[k] = c;
    }
  }
  __syncthreads();

  if (t < INN) {
    float s = 0.f;
    for (int p = 0; p < PP; ++p) s += As[t][p] * cf[p];
    out_xhat[(size_t)b * INN + t] = s;          // SCALE=1, SHIFT=0
  }
  if (t >= INN && t < INN + PP)
    out_coeff[(size_t)b * PP + (t - INN)] = cf[t - INN];
}

// ---------------------------------------------------------------------------
// Host orchestration (graph-capture safe: only kernel launches on stream)
// ---------------------------------------------------------------------------
extern "C" void kernel_launch(void* const* d_in, const int* in_sizes, int n_in,
                              void* d_out, int out_size, void* d_ws,
                              size_t ws_size, hipStream_t stream) {
  (void)in_sizes; (void)n_in; (void)out_size; (void)ws_size;

  const float* x      = (const float*)d_in[0];
  const float* enc_w1 = (const float*)d_in[1];
  const float* enc_b1 = (const float*)d_in[2];
  const float* res_w  = (const float*)d_in[3];
  const float* res_b  = (const float*)d_in[4];
  const float* res_g  = (const float*)d_in[5];
  const float* res_be = (const float*)d_in[6];
  const float* enc_w2 = (const float*)d_in[7];
  const float* enc_b2 = (const float*)d_in[8];
  const float* emb    = (const float*)d_in[9];

  // workspace layout (~131 MB), 256B-aligned sections
  char* base = (char*)d_ws;
  size_t off = 0;
  auto alloc = [&](size_t bytes) {
    void* p = base + off;
    off = (off + bytes + 255) & ~(size_t)255;
    return p;
  };
  float*     hbuf    = (float*)alloc((size_t)BB * DENC * 4);
  float*     zbuf    = (float*)alloc((size_t)BB * ZD * 4);
  int*       idxi    = (int*)alloc((size_t)BP * 4);
  float*     counts  = (float*)alloc((size_t)NE * 4);
  float*     en2h    = (float*)alloc((size_t)NE * 4);
  float*     lossacc = (float*)alloc(4);
  _Float16*  xh      = (_Float16*)alloc((size_t)BB * INN * 2);
  _Float16*  hh      = (_Float16*)alloc((size_t)BB * DENC * 2);
  _Float16*  zh      = (_Float16*)alloc((size_t)BB * ZD * 2);
  _Float16*  w1t     = (_Float16*)alloc((size_t)DENC * INN * 2);
  _Float16*  wrt     = (_Float16*)alloc((size_t)DENC * DENC * 2);
  _Float16*  w2t     = (_Float16*)alloc((size_t)ZD * DENC * 2);
  _Float16*  embh    = (_Float16*)alloc((size_t)NE * INN * 2);

  // output layout: loss | x_hat(BB*INN) | perplexity | idx(BP) | coeff(BP)
  float* out       = (float*)d_out;
  float* out_loss  = out;
  float* out_xhat  = out + 1;
  float* out_perp  = out + 1 + (size_t)BB * INN;
  float* out_idxf  = out_perp + 1;
  float* out_coeff = out_idxf + BP;

  convert_kernel<<<512, 256, 0, stream>>>(x, enc_w1, res_w, enc_w2, emb, xh,
                                          w1t, wrt, w2t, embh, en2h, counts,
                                          lossacc);

  // h = x @ W1 + b1      (M=16384, N=512, K=64)
  gemm_bias_kernel<<<dim3(BB / 16, DENC / 128), 256, 0, stream>>>(
      xh, w1t, enc_b1, hbuf, hh, DENC, INN);

  // 2 shared-weight residual blocks (fused GEMM + LN + ReLU + add)
  for (int it = 0; it < 2; ++it)
    resblock_kernel<<<BB / 16, 256, 0, stream>>>(hbuf, hh, wrt, res_b, res_g,
                                                 res_be);

  // z = h @ W2 + b2      (M=16384, N=768, K=512)
  gemm_bias_kernel<<<dim3(BB / 16, ZD / 128), 256, 0, stream>>>(
      hh, w2t, enc_b2, zbuf, zh, ZD, DENC);

  // nearest-codeword search (async-staged codebook + WMMA distance GEMM)
  argmin_kernel<<<BP / 16, 256, 0, stream>>>(zh, embh, en2h, idxi, out_idxf);

  // VQ loss + usage histogram
  loss_counts_kernel<<<768, 256, 0, stream>>>(zbuf, emb, idxi, lossacc, counts);

  // scalars
  finalize_kernel<<<1, 256, 0, stream>>>(lossacc, counts, out_loss, out_perp);

  // per-block least-squares reconstruction
  pinv_kernel<<<BB, 128, 0, stream>>>(x, emb, idxi, out_xhat, out_coeff);
}
